// Decoder_51582557225708
// MI455X (gfx1250) — matrix operand order repaired and checked by reading, not compiled, not run on hardware
//
#include <hip/hip_runtime.h>

typedef _Float16 half_t;
typedef __attribute__((ext_vector_type(8)))  _Float16 v8h;
typedef __attribute__((ext_vector_type(16))) _Float16 v16h;
typedef __attribute__((ext_vector_type(8)))  float    v8f;

#define D_MODEL 2048
#define SEQ     1024
#define NH      16
#define HD      128
#define CACHE   2048
#define KTOT    3072
#define DFF     5632
#define NLAYER  2
#define CHUNK   512

__device__ __forceinline__ v16h cat8(v8h a, v8h b){
  v16h r;
#pragma unroll
  for (int i=0;i<8;++i){ r[i]=a[i]; r[i+8]=b[i]; }
  return r;
}
__device__ __forceinline__ v8f wmma_f16(v16h a, v16h b, v8f c){
  return __builtin_amdgcn_wmma_f32_16x16x32_f16(false, a, false, b, (short)0, c, false, false);
}

// ---------------- GEMM: C[M,N] = A[M,K](f16,row) * Bt[N,K](f16,row)^T (+res) ----------------
// Workgroup tile 128x256, 8 waves as 2(m) x 4(n), wave tile 64x64 (16 WMMA / 8 frag-loads per K-step).
// Double-buffered LDS, one barrier per K-step.
#define BM 128
#define BN 256
#define BK 32
#define LDT 40

__global__ __launch_bounds__(256) void gemm_f16f32(
    const half_t* __restrict__ A, const half_t* __restrict__ Bt,
    const float* __restrict__ res, float* __restrict__ C,
    int M, int N, int K)
{
  __shared__ __align__(16) half_t sA[2][BM*LDT];
  __shared__ __align__(16) half_t sB[2][BN*LDT];
  const int t    = threadIdx.x;
  const int lane = t & 31;
  const int wv   = t >> 5;
  const int wm   = wv >> 2;      // 0..1
  const int wn   = wv & 3;       // 0..3
  const int ln   = lane & 15;
  const int hi   = lane >> 4;
  const int m0   = blockIdx.y * BM;
  const int n0   = blockIdx.x * BN;

  const int arow = t >> 1;         // 0..127
  const int acol = (t & 1) * 16;   // 0 or 16
  const half_t* ga = A  + (size_t)(m0 + arow) * K + acol;
  const half_t* gb = Bt + (size_t)(n0 + t) * K;     // one 256-row B tile: row per thread

  v8f acc[4][4];
#pragma unroll
  for (int i=0;i<4;++i)
#pragma unroll
    for (int j=0;j<4;++j)
#pragma unroll
      for (int e=0;e<8;++e) acc[i][j][e] = 0.0f;

  v8h ra0 = *(const v8h*)(ga);
  v8h ra1 = *(const v8h*)(ga + 8);
  v8h rb[4];
#pragma unroll
  for (int j=0;j<4;++j) rb[j] = *(const v8h*)(gb + j*8);

  // prime buffer 0
  *(v8h*)(sA[0] + arow*LDT + acol)     = ra0;
  *(v8h*)(sA[0] + arow*LDT + acol + 8) = ra1;
#pragma unroll
  for (int j=0;j<4;++j) *(v8h*)(sB[0] + t*LDT + j*8) = rb[j];

  const int nk = K / BK;
  for (int kt = 0; kt < nk; ++kt){
    const int p = kt & 1;
    const bool more = (kt + 1) < nk;
    if (more){
      ra0 = *(const v8h*)(ga + (kt+1)*BK);
      ra1 = *(const v8h*)(ga + (kt+1)*BK + 8);
#pragma unroll
      for (int j=0;j<4;++j) rb[j] = *(const v8h*)(gb + (kt+1)*BK + j*8);
    }
    if (kt + 2 < nk){
      __builtin_prefetch(ga + (kt+2)*BK, 0, 1);
      __builtin_prefetch(gb + (kt+2)*BK, 0, 1);
    }
    __syncthreads();
    v16h bf[4];
#pragma unroll
    for (int kn=0;kn<4;++kn){
      const half_t* pB = sB[p] + (wn*64 + kn*16 + ln)*LDT + 8*hi;
      bf[kn] = cat8(*(const v8h*)pB, *(const v8h*)(pB + 16));
    }
#pragma unroll
    for (int km=0;km<4;++km){
      const half_t* pA = sA[p] + (wm*64 + km*16 + ln)*LDT + 8*hi;
      v16h af = cat8(*(const v8h*)pA, *(const v8h*)(pA + 16));
#pragma unroll
      for (int kn=0;kn<4;++kn)
        acc[km][kn] = wmma_f16(af, bf[kn], acc[km][kn]);
    }
    if (more){
      *(v8h*)(sA[1-p] + arow*LDT + acol)     = ra0;
      *(v8h*)(sA[1-p] + arow*LDT + acol + 8) = ra1;
#pragma unroll
      for (int j=0;j<4;++j) *(v8h*)(sB[1-p] + t*LDT + j*8) = rb[j];
    }
  }

#pragma unroll
  for (int km=0;km<4;++km)
#pragma unroll
    for (int kn=0;kn<4;++kn)
#pragma unroll
      for (int j=0;j<8;++j){
        int r = m0 + wm*64 + km*16 + j + 8*hi;
        int c = n0 + wn*64 + kn*16 + ln;
        float v = acc[km][kn][j];
        if (res) v += res[(size_t)r*N + c];
        C[(size_t)r*N + c] = v;
      }
}

// ---------------- Flash attention: per (head, 128-q-row tile) ----------------
#define AKB 64
#define SKL 136
#define SVL 72

__global__ __launch_bounds__(256) void attn_fa(
    const half_t* __restrict__ Q, const half_t* __restrict__ Kf,
    const half_t* __restrict__ Vt, half_t* __restrict__ O16)
{
  __shared__ __align__(16) half_t sK[AKB*SKL];     // [key][dim]
  __shared__ __align__(16) half_t sV[HD*SVL];      // [dim][key]
  __shared__ __align__(16) half_t sP[8*16*SVL];    // per-wave P tile
  const int h  = blockIdx.x;
  const int qt = blockIdx.y;
  const int t  = threadIdx.x;
  const int lane = t & 31;
  const int wv   = t >> 5;
  const int ln   = lane & 15;
  const int hi   = lane >> 4;

  const half_t* qbase = Q + ((size_t)h*SEQ + qt*128 + wv*16 + ln) * HD;
  v16h qf[4];
#pragma unroll
  for (int ck=0;ck<4;++ck){
    const half_t* p = qbase + ck*32 + 8*hi;
    qf[ck] = cat8(*(const v8h*)p, *(const v8h*)(p + 16));
  }

  v8f o[8];
  float mrow[8], lrow[8];
#pragma unroll
  for (int v2=0;v2<8;++v2)
#pragma unroll
    for (int e=0;e<8;++e) o[v2][e] = 0.0f;
#pragma unroll
  for (int j=0;j<8;++j){ mrow[j] = -3.0e38f; lrow[j] = 0.0f; }

  const int nloc = 2*(qt & 3) + 2;
  const int nblk = CACHE/AKB + nloc;
  const int cb   = CACHE + (qt >> 2) * CHUNK;
  half_t* pw = sP + wv*16*SVL;

  for (int bi = 0; bi < nblk; ++bi){
    const int kbase = (bi < CACHE/AKB) ? bi*AKB : cb + (bi - CACHE/AKB)*AKB;
    const bool loc = bi >= CACHE/AKB;
    __syncthreads();
#pragma unroll
    for (int it=0; it<4; ++it){
      int slot = t + it*256;
      int r = slot >> 4, c8 = (slot & 15) * 8;
      *(v8h*)(sK + r*SKL + c8) = *(const v8h*)(Kf + ((size_t)h*KTOT + kbase + r)*HD + c8);
    }
#pragma unroll
    for (int it=0; it<4; ++it){
      int slot = t + it*256;
      int r = slot >> 3, c8 = (slot & 7) * 8;
      *(v8h*)(sV + r*SVL + c8) = *(const v8h*)(Vt + ((size_t)h*HD + r)*KTOT + kbase + c8);
    }
    __syncthreads();

    v8f s[4];
#pragma unroll
    for (int tk=0;tk<4;++tk){
#pragma unroll
      for (int e=0;e<8;++e) s[tk][e] = 0.0f;
#pragma unroll
      for (int ck=0;ck<4;++ck){
        const half_t* p = sK + (tk*16 + ln)*SKL + ck*32 + 8*hi;
        v16h bf = cat8(*(const v8h*)p, *(const v8h*)(p + 16));
        s[tk] = wmma_f16(qf[ck], bf, s[tk]);
      }
    }
    const float sc = 0.088388347648318f; // 1/sqrt(128)
#pragma unroll
    for (int tk=0;tk<4;++tk)
#pragma unroll
      for (int j=0;j<8;++j){
        float v = s[tk][j] * sc;
        if (loc){
          int jl = kbase - CACHE + tk*16 + ln;
          int qi = qt*128 + wv*16 + j + 8*hi;
          if (jl > qi) v = -3.0e38f;
        }
        s[tk][j] = v;
      }
    float mb[8], al[8], ps[8];
#pragma unroll
    for (int j=0;j<8;++j){
      mb[j] = fmaxf(fmaxf(s[0][j], s[1][j]), fmaxf(s[2][j], s[3][j]));
      mb[j] = fmaxf(mb[j], __shfl_xor(mb[j], 8));
      mb[j] = fmaxf(mb[j], __shfl_xor(mb[j], 4));
      mb[j] = fmaxf(mb[j], __shfl_xor(mb[j], 2));
      mb[j] = fmaxf(mb[j], __shfl_xor(mb[j], 1));
      float mn = fmaxf(mrow[j], mb[j]);
      al[j] = __expf(mrow[j] - mn);
      mrow[j] = mn;
    }
#pragma unroll
    for (int tk=0;tk<4;++tk)
#pragma unroll
      for (int j=0;j<8;++j)
        s[tk][j] = __expf(s[tk][j] - mrow[j]);
#pragma unroll
    for (int j=0;j<8;++j){
      ps[j] = s[0][j] + s[1][j] + s[2][j] + s[3][j];
      ps[j] += __shfl_xor(ps[j], 8);
      ps[j] += __shfl_xor(ps[j], 4);
      ps[j] += __shfl_xor(ps[j], 2);
      ps[j] += __shfl_xor(ps[j], 1);
      lrow[j] = lrow[j]*al[j] + ps[j];
    }
#pragma unroll
    for (int v2=0;v2<8;++v2)
#pragma unroll
      for (int j=0;j<8;++j) o[v2][j] *= al[j];
#pragma unroll
    for (int tk=0;tk<4;++tk)
#pragma unroll
      for (int j=0;j<8;++j)
        pw[(j + 8*hi)*SVL + tk*16 + ln] = (half_t)s[tk][j];
#pragma unroll
    for (int kc=0;kc<2;++kc){
      const half_t* pa = pw + ln*SVL + kc*32 + 8*hi;
      v16h af = cat8(*(const v8h*)pa, *(const v8h*)(pa + 16));
#pragma unroll
      for (int v2=0;v2<8;++v2){
        const half_t* pb = sV + (v2*16 + ln)*SVL + kc*32 + 8*hi;
        v16h bf = cat8(*(const v8h*)pb, *(const v8h*)(pb + 16));
        o[v2] = wmma_f16(af, bf, o[v2]);
      }
    }
  }
#pragma unroll
  for (int j=0;j<8;++j) lrow[j] = 1.0f / lrow[j];
#pragma unroll
  for (int v2=0;v2<8;++v2)
#pragma unroll
    for (int j=0;j<8;++j){
      size_t r = (size_t)(qt*128 + wv*16 + j + 8*hi);
      O16[r*D_MODEL + h*HD + v2*16 + ln] = (half_t)(o[v2][j] * lrow[j]);
    }
}

// ---------------- small kernels ----------------
__global__ __launch_bounds__(256) void transpose_cvt(const float* __restrict__ src,
                                                     half_t* __restrict__ dst, int K, int N){
  __shared__ float tile[32][33];
  int nt = blockIdx.x*32, kt = blockIdx.y*32;
  int tx = threadIdx.x & 31, ty = threadIdx.x >> 5;
#pragma unroll
  for (int i=0;i<4;++i)
    tile[ty + i*8][tx] = src[(size_t)(kt + ty + i*8)*N + nt + tx];
  __syncthreads();
#pragma unroll
  for (int i=0;i<4;++i)
    dst[(size_t)(nt + ty + i*8)*K + kt + tx] = (half_t)tile[tx][ty + i*8];
}

__global__ __launch_bounds__(256) void embed_kernel(const int* __restrict__ ids,
                                                    const float* __restrict__ emb,
                                                    float* __restrict__ x){
  int idx = blockIdx.x*256 + threadIdx.x;
  int s = idx >> 11, d = idx & 2047;
  x[idx] = emb[(size_t)ids[s]*D_MODEL + d];
}

__global__ __launch_bounds__(256) void rmsnorm_kernel(const float* __restrict__ x,
                                                      const float* __restrict__ w,
                                                      half_t* __restrict__ out){
  __shared__ float red[8];
  int row = blockIdx.x;
  const float* xr = x + (size_t)row*D_MODEL;
  float ss = 0.0f;
  for (int i = threadIdx.x; i < D_MODEL; i += 256){ float v = xr[i]; ss += v*v; }
  ss += __shfl_xor(ss,16); ss += __shfl_xor(ss,8); ss += __shfl_xor(ss,4);
  ss += __shfl_xor(ss,2);  ss += __shfl_xor(ss,1);
  if ((threadIdx.x & 31) == 0) red[threadIdx.x >> 5] = ss;
  __syncthreads();
  float tot = 0.0f;
#pragma unroll
  for (int i=0;i<8;++i) tot += red[i];
  float inv = rsqrtf(tot * (1.0f/D_MODEL) + 1e-5f);
  half_t* orow = out + (size_t)row*D_MODEL;
  for (int i = threadIdx.x; i < D_MODEL; i += 256) orow[i] = (half_t)(xr[i]*inv*w[i]);
}

__global__ __launch_bounds__(256) void rope_q_kernel(const float* __restrict__ qkv,
                                                     half_t* __restrict__ Qf){
  int idx = blockIdx.x*256 + threadIdx.x;   // SEQ*D_MODEL
  int s = idx >> 11, col = idx & 2047;
  int hh = col >> 7, d = col & 127;
  int pos = CACHE + s;
  int d2 = d & 63;
  float inv = __powf(10000.0f, -(float)(2*d2) * (1.0f/128.0f));
  float fr = (float)pos * inv;
  float c = cosf(fr), sn = sinf(fr);
  float x0 = qkv[(size_t)s*3072 + col];
  float xr = (d < 64) ? -qkv[(size_t)s*3072 + col + 64] : qkv[(size_t)s*3072 + col - 64];
  Qf[((size_t)hh*SEQ + s)*HD + d] = (half_t)(x0*c + xr*sn);
}

__global__ __launch_bounds__(256) void prep_k_kernel(const float* __restrict__ kv,
                                                     const float* __restrict__ qkv,
                                                     half_t* __restrict__ Kf, int l){
  int idx = blockIdx.x*256 + threadIdx.x;   // NH*KTOT*HD
  int d = idx & 127;
  int rem = idx >> 7;
  int pos = rem % KTOT;
  int hh  = rem / KTOT;
  float x0, xr;
  if (pos < CACHE){
    size_t base = (((size_t)l*NH + hh)*CACHE + pos)*(size_t)HD;   // k-part [0][l][0][hh][pos][:]
    x0 = kv[base + d];
    xr = (d < 64) ? -kv[base + d + 64] : kv[base + d - 64];
  } else {
    int s = pos - CACHE;
    size_t base = (size_t)s*3072 + 2048 + (size_t)(hh >> 2)*HD;
    x0 = qkv[base + d];
    xr = (d < 64) ? -qkv[base + d + 64] : qkv[base + d - 64];
  }
  int d2 = d & 63;
  float inv = __powf(10000.0f, -(float)(2*d2) * (1.0f/128.0f));
  float fr = (float)pos * inv;
  float c = cosf(fr), sn = sinf(fr);
  Kf[((size_t)hh*KTOT + pos)*HD + d] = (half_t)(x0*c + xr*sn);
}

__global__ __launch_bounds__(256) void prep_v_kernel(const float* __restrict__ kv,
                                                     const float* __restrict__ qkv,
                                                     half_t* __restrict__ Vt, int l){
  int idx = blockIdx.x*256 + threadIdx.x;   // NH*HD*KTOT, pos fastest
  int pos = idx % KTOT;
  int rem = idx / KTOT;
  int d  = rem & 127;
  int hh = rem >> 7;
  float v;
  if (pos < CACHE){
    v = kv[(((size_t)(NLAYER + l)*NH + hh)*CACHE + pos)*(size_t)HD + d];  // v-part [1][l][0][hh][pos][d]
  } else {
    int s = pos - CACHE;
    v = qkv[(size_t)s*3072 + 2560 + (size_t)(hh >> 2)*HD + d];
  }
  Vt[((size_t)hh*HD + d)*KTOT + pos] = (half_t)v;
}

__global__ __launch_bounds__(256) void swiglu_kernel(const float* __restrict__ gu,
                                                     half_t* __restrict__ out){
  int idx = blockIdx.x*256 + threadIdx.x;
  if (idx >= SEQ*DFF) return;
  int s = idx / DFF, f = idx - s*DFF;
  float g = gu[(size_t)s*(2*DFF) + f];
  float u = gu[(size_t)s*(2*DFF) + DFF + f];
  float sg = g / (1.0f + __expf(-g));
  out[idx] = (half_t)(sg*u);
}

// ---------------- host ----------------
extern "C" void kernel_launch(void* const* d_in, const int* in_sizes, int n_in,
                              void* d_out, int out_size, void* d_ws, size_t ws_size,
                              hipStream_t stream){
  const int*   ids = (const int*)  d_in[0];
  const float* kv  = (const float*)d_in[1];
  const float* emb = (const float*)d_in[2];
  const float* Wq  = (const float*)d_in[3];
  const float* Wk  = (const float*)d_in[4];
  const float* Wv  = (const float*)d_in[5];
  const float* Wo  = (const float*)d_in[6];
  const float* ln1 = (const float*)d_in[7];
  const float* ln2 = (const float*)d_in[8];
  const float* Wg  = (const float*)d_in[9];
  const float* Wu  = (const float*)d_in[10];
  const float* Wd  = (const float*)d_in[11];
  const float* nw  = (const float*)d_in[12];
  const float* lmh = (const float*)d_in[13];
  float* out = (float*)d_out;

  char* base = (char*)d_ws;
  size_t off = 0;
  auto alloc = [&](size_t bytes)->void*{
    void* p = base + off;
    off += (bytes + 255) & ~(size_t)255;
    return p;
  };
  half_t *qkvt[NLAYER], *wot[NLAYER], *wgut[NLAYER], *wdt[NLAYER];
  for (int l=0;l<NLAYER;++l){
    qkvt[l] = (half_t*)alloc((size_t)3072*2048*2);
    wot[l]  = (half_t*)alloc((size_t)2048*2048*2);
    wgut[l] = (half_t*)alloc((size_t)11264*2048*2);
    wdt[l]  = (half_t*)alloc((size_t)2048*5632*2);
  }
  half_t* lmt  = (half_t*)alloc((size_t)32000*2048*2);
  float*  xa   = (float*) alloc((size_t)SEQ*D_MODEL*4);
  float*  xb   = (float*) alloc((size_t)SEQ*D_MODEL*4);
  half_t* h16  = (half_t*)alloc((size_t)SEQ*D_MODEL*2);
  float*  qkvf = (float*) alloc((size_t)SEQ*3072*4);
  half_t* Qf   = (half_t*)alloc((size_t)NH*SEQ*HD*2);
  half_t* Kf   = (half_t*)alloc((size_t)NH*KTOT*HD*2);
  half_t* Vtb  = (half_t*)alloc((size_t)NH*HD*KTOT*2);
  half_t* att  = (half_t*)alloc((size_t)SEQ*D_MODEL*2);
  float*  gu   = (float*) alloc((size_t)SEQ*2*DFF*4);
  half_t* act  = (half_t*)alloc((size_t)SEQ*DFF*2);

  for (int l=0;l<NLAYER;++l){
    transpose_cvt<<<dim3(64, 64),  256, 0, stream>>>(Wq + (size_t)l*2048*2048, qkvt[l],                     2048, 2048);
    transpose_cvt<<<dim3(16, 64),  256, 0, stream>>>(Wk + (size_t)l*2048*512,  qkvt[l] + (size_t)2048*2048, 2048, 512);
    transpose_cvt<<<dim3(16, 64),  256, 0, stream>>>(Wv + (size_t)l*2048*512,  qkvt[l] + (size_t)2560*2048, 2048, 512);
    transpose_cvt<<<dim3(64, 64),  256, 0, stream>>>(Wo + (size_t)l*2048*2048, wot[l],                      2048, 2048);
    transpose_cvt<<<dim3(176, 64), 256, 0, stream>>>(Wg + (size_t)l*2048*5632, wgut[l],                     2048, 5632);
    transpose_cvt<<<dim3(176, 64), 256, 0, stream>>>(Wu + (size_t)l*2048*5632, wgut[l] + (size_t)5632*2048, 2048, 5632);
    transpose_cvt<<<dim3(64, 176), 256, 0, stream>>>(Wd + (size_t)l*5632*2048, wdt[l],                      5632, 2048);
  }
  transpose_cvt<<<dim3(1000, 64), 256, 0, stream>>>(lmh, lmt, 2048, 32000);

  embed_kernel<<<SEQ*D_MODEL/256, 256, 0, stream>>>(ids, emb, xa);

  float* x = xa; float* x2 = xb;
  for (int l=0;l<NLAYER;++l){
    rmsnorm_kernel<<<SEQ, 256, 0, stream>>>(x, ln1 + (size_t)l*D_MODEL, h16);
    gemm_f16f32<<<dim3(3072/BN, SEQ/BM), 256, 0, stream>>>(h16, qkvt[l], nullptr, qkvf, SEQ, 3072, 2048);
    rope_q_kernel<<<SEQ*D_MODEL/256, 256, 0, stream>>>(qkvf, Qf);
    prep_k_kernel<<<NH*KTOT*HD/256, 256, 0, stream>>>(kv, qkvf, Kf, l);
    prep_v_kernel<<<NH*KTOT*HD/256, 256, 0, stream>>>(kv, qkvf, Vtb, l);
    attn_fa<<<dim3(NH, SEQ/128), 256, 0, stream>>>(Qf, Kf, Vtb, att);
    gemm_f16f32<<<dim3(2048/BN, SEQ/BM), 256, 0, stream>>>(att, wot[l], x, x2, SEQ, 2048, 2048);
    rmsnorm_kernel<<<SEQ, 256, 0, stream>>>(x2, ln2 + (size_t)l*D_MODEL, h16);
    gemm_f16f32<<<dim3(11264/BN, SEQ/BM), 256, 0, stream>>>(h16, wgut[l], nullptr, gu, SEQ, 11264, 2048);
    swiglu_kernel<<<(SEQ*DFF + 255)/256, 256, 0, stream>>>(gu, act);
    gemm_f16f32<<<dim3(2048/BN, SEQ/BM), 256, 0, stream>>>(act, wdt[l], x2, x, SEQ, 2048, 5632);
  }
  rmsnorm_kernel<<<SEQ, 256, 0, stream>>>(x, nw, h16);
  gemm_f16f32<<<dim3(32000/BN, SEQ/BM), 256, 0, stream>>>(h16, lmt, nullptr, out, SEQ, 32000, 2048);

  (void)in_sizes; (void)n_in; (void)out_size; (void)ws_size;
}
